// SingleLayerCTRNN_73100343378141
// MI455X (gfx1250) — matrix operand order repaired and checked by reading, not compiled, not run on hardware
//
#include <hip/hip_runtime.h>

// ---------------------------------------------------------------------------
// CTRNN on MI455X (gfx1250, wave32):
//   T=512 serial steps; N=256 batch split as 16 workgroups x 16 rows.
//   Per WG: h state (16x512 f32) + A=nonlin(h) (16x512 f16) resident in LDS.
//   Recurrent/input/output GEMMs via v_wmma_f32_16x16x32_f16, f32 accumulate.
//   Weights pre-transposed to column-major f16 in d_ws (B-fragment friendly).
//   Nonlinearity relu(tanh(x)) done branch-free with native TRANS ops.
// ---------------------------------------------------------------------------

typedef __attribute__((ext_vector_type(16))) _Float16 v16h;
typedef __attribute__((ext_vector_type(8)))  _Float16 v8h;
typedef __attribute__((ext_vector_type(8)))  float    v8f;

#define HDIM 512
#define TSTEPS 512
#define NBATCH 256
#define DIN 64
#define DOUT 64

union F16x16 { v16h v; v8h h[2]; };

__device__ __forceinline__ v8f wmma16(v16h a, v16h b, v8f c) {
  // D = A(16x32 f16) * B(32x16 f16) + C(16x16 f32)
  return __builtin_amdgcn_wmma_f32_16x16x32_f16(
      /*neg_a=*/false, a, /*neg_b=*/false, b,
      /*c_mod=*/(short)0, c, /*reuse_a=*/false, /*reuse_b=*/false);
}

// A-fragment (16x32 f16) from row-major buffer (stride ld halves), tile at k0.
// ISA layout: lane<16 -> row=lane, K=[k0..k0+7] in v0-3, K=[k0+16..k0+23] in v4-7
//             lane>=16 -> row=lane-16, K offsets +8.
__device__ __forceinline__ v16h load_a_frag(const _Float16* base, int ld,
                                            int k0, int lane) {
  const int m   = lane & 15;
  const int sel = lane >> 4;
  const _Float16* p = base + m * ld + k0 + sel * 8;
  F16x16 u;
  u.h[0] = *(const v8h*)(p);        // 16B -> VGPR0-3
  u.h[1] = *(const v8h*)(p + 16);   // 16B -> VGPR4-7
  return u.v;
}

// B-fragment (32x16 f16) from column-major weight Wt[n*ldk + k], tile (k0,n0).
// ISA layout: lane<16 -> K=[k0..k0+7],[k0+16..k0+23] of col n=lane; lane>=16 -> K offsets +8.
__device__ __forceinline__ v16h load_b_frag(const _Float16* W, int ldk,
                                            int n0, int k0, int lane) {
  const int n   = n0 + (lane & 15);
  const int sel = lane >> 4;
  const _Float16* p = W + (size_t)n * ldk + k0 + sel * 8;
  F16x16 u;
  u.h[0] = *(const v8h*)(p);        // K[8*sel..8*sel+7]
  u.h[1] = *(const v8h*)(p + 16);   // K[16+8*sel..16+8*sel+7]
  return u.v;
}

// relu(tanh(x)) without branches or libm:
//   tanh is monotone & sign-preserving, so relu(tanh(x)) = tanh(max(x,0)).
//   For m >= 0: tanh(m) = 1 - 2/(e^{2m}+1), e^{2m} = exp2(m * 2*log2(e)).
//   Limits: m=0 -> 1-2*rcp(2)=0; m->inf -> exp2->inf, rcp->0, result->1.
__device__ __forceinline__ float nlin(float x) {
  const float m = fmaxf(x, 0.0f);
#if defined(__HIP_DEVICE_COMPILE__)
#if __has_builtin(__builtin_amdgcn_tanhf)
  return __builtin_amdgcn_tanhf(m);          // native v_tanh_f32 (gfx1250)
#elif __has_builtin(__builtin_amdgcn_exp2f) && __has_builtin(__builtin_amdgcn_rcpf)
  const float e = __builtin_amdgcn_exp2f(m * 2.8853900817779268f);
  return 1.0f - 2.0f * __builtin_amdgcn_rcpf(e + 1.0f);
#else
  const float e = exp2f(m * 2.8853900817779268f);
  return 1.0f - 2.0f / (e + 1.0f);
#endif
#else
  return m;  // host pass: never executed, just needs to parse/compile
#endif
}

// ---- prep: dst[n*K + k] = (f16) src[k*N + n], K a power of two ----
__global__ __launch_bounds__(256) void transpose_to_f16(
    const float* __restrict__ src, _Float16* __restrict__ dst,
    int logK, int total, int N) {
  int i = blockIdx.x * blockDim.x + threadIdx.x;
  if (i >= total) return;
  int K = 1 << logK;
  int n = i >> logK;
  int k = i & (K - 1);
  dst[i] = (_Float16)src[(size_t)k * N + n];
}

__global__ __launch_bounds__(256) void ctrnn_kernel(
    const float* __restrict__ h0_data, const float* __restrict__ input_ts,
    const float* __restrict__ b_h0, const float* __restrict__ b_rec,
    const float* __restrict__ alpha,
    const _Float16* __restrict__ Wt_h0, const _Float16* __restrict__ Wt_in,
    const _Float16* __restrict__ Wt_rec, const _Float16* __restrict__ Wt_out,
    float* __restrict__ out) {
  __shared__ alignas(16) float    h_lds[16 * HDIM];   // 32 KB f32 state
  __shared__ alignas(16) _Float16 a_lds[16 * HDIM];   // 16 KB f16 nonlin(h)
  __shared__ alignas(16) _Float16 x_lds[16 * DIN];    //  2 KB f16 x_t tile

  const int tid  = threadIdx.x;
  const int wave = tid >> 5;           // 8 waves; wave owns H cols [64w,64w+64)
  const int lane = tid & 31;
  const int row0 = blockIdx.x * 16;    // batch rows [row0, row0+16)
  const int sel  = lane >> 4;
  const int ncol = lane & 15;          // C-frag column within tile
  const int mbase = sel * 8;           // C-frag row base (0 or 8)

  const float av = alpha[0];
  const float om = 1.0f - av;

  float brec[4];
#pragma unroll
  for (int j = 0; j < 4; ++j) brec[j] = b_rec[wave * 64 + j * 16 + ncol];

  // -------- h0 = h0_data @ W_h0 + b_h0 --------
  for (int i = tid; i < 16 * DIN; i += 256)
    x_lds[i] = (_Float16)h0_data[(size_t)(row0 + (i >> 6)) * DIN + (i & 63)];
  __syncthreads();

  {
    v8f acc[4] = {{}, {}, {}, {}};
#pragma unroll
    for (int kt = 0; kt < 2; ++kt) {
      v16h a = load_a_frag(x_lds, DIN, kt * 32, lane);
#pragma unroll
      for (int j = 0; j < 4; ++j) {
        v16h b = load_b_frag(Wt_h0, DIN, wave * 64 + j * 16, kt * 32, lane);
        acc[j] = wmma16(a, b, acc[j]);
      }
    }
#pragma unroll
    for (int j = 0; j < 4; ++j) {
      const int n0 = wave * 64 + j * 16;
      const float bb = b_h0[n0 + ncol];
#pragma unroll
      for (int r = 0; r < 8; ++r) {
        const float hv = acc[j][r] + bb;
        const int idx = (mbase + r) * HDIM + n0 + ncol;
        h_lds[idx] = hv;
        a_lds[idx] = (_Float16)nlin(hv);
      }
    }
  }
  __syncthreads();

  // -------- out[0] = nonlin(h0) @ W_out (waves 0-3, one 16x16 tile each) ----
  if (wave < 4) {
    const int n0 = wave * 16;
    v8f o = {};
#pragma unroll
    for (int kt = 0; kt < 16; ++kt) {
      v16h a = load_a_frag(a_lds, HDIM, kt * 32, lane);
      v16h b = load_b_frag(Wt_out, HDIM, n0, kt * 32, lane);
      o = wmma16(a, b, o);
    }
#pragma unroll
    for (int r = 0; r < 8; ++r)
      out[((size_t)0 * NBATCH + row0 + mbase + r) * DOUT + n0 + ncol] = o[r];
  }

  // -------- serial scan over t = 1..T-1 --------
  for (int t = 1; t < TSTEPS; ++t) {
    // stage x_t tile -> f16 LDS
    const float* xs = input_ts + (size_t)t * NBATCH * DIN + (size_t)row0 * DIN;
    for (int i = tid; i < 16 * DIN; i += 256) x_lds[i] = (_Float16)xs[i];
    __syncthreads();  // B1: x staged; a_lds stable from previous step

    // acc = nonlin(h_prev) @ W_rec + x_t @ W_in
    // k-outer / n-tile-inner: one A fragment (2x ds_load_b128) feeds 4 WMMAs.
    v8f acc[4] = {{}, {}, {}, {}};
#pragma unroll
    for (int kt = 0; kt < 16; ++kt) {
      v16h a = load_a_frag(a_lds, HDIM, kt * 32, lane);
#pragma unroll
      for (int j = 0; j < 4; ++j) {
        v16h b = load_b_frag(Wt_rec, HDIM, wave * 64 + j * 16, kt * 32, lane);
        acc[j] = wmma16(a, b, acc[j]);
      }
    }
#pragma unroll
    for (int kt = 0; kt < 2; ++kt) {
      v16h a = load_a_frag(x_lds, DIN, kt * 32, lane);
#pragma unroll
      for (int j = 0; j < 4; ++j) {
        v16h b = load_b_frag(Wt_in, DIN, wave * 64 + j * 16, kt * 32, lane);
        acc[j] = wmma16(a, b, acc[j]);
      }
    }
    __syncthreads();  // B2: all reads of a_lds done; safe to overwrite

    // leaky update + write-back of h and A = f16(nonlin(h))
#pragma unroll
    for (int j = 0; j < 4; ++j) {
      const int n0 = wave * 64 + j * 16;
#pragma unroll
      for (int r = 0; r < 8; ++r) {
        const int idx = (mbase + r) * HDIM + n0 + ncol;
        const float hp = h_lds[idx];
        const float hv = om * hp + av * (acc[j][r] + brec[j]);
        h_lds[idx] = hv;
        a_lds[idx] = (_Float16)nlin(hv);
      }
    }
    __syncthreads();  // B3: new a_lds visible

    // out[t] = nonlin(h_t) @ W_out  (waves 0-3; waves 4-7 run ahead to stage x)
    if (wave < 4) {
      const int n0 = wave * 16;
      v8f o = {};
#pragma unroll
      for (int kt = 0; kt < 16; ++kt) {
        v16h a = load_a_frag(a_lds, HDIM, kt * 32, lane);
        v16h b = load_b_frag(Wt_out, HDIM, n0, kt * 32, lane);
        o = wmma16(a, b, o);
      }
#pragma unroll
      for (int r = 0; r < 8; ++r)
        out[((size_t)t * NBATCH + row0 + mbase + r) * DOUT + n0 + ncol] = o[r];
    }
  }
}

extern "C" void kernel_launch(void* const* d_in, const int* in_sizes, int n_in,
                              void* d_out, int out_size, void* d_ws,
                              size_t ws_size, hipStream_t stream) {
  const float* h0_data  = (const float*)d_in[0];  // (1,256,64)
  const float* input_ts = (const float*)d_in[1];  // (512,256,64)
  const float* W_h0     = (const float*)d_in[2];  // (64,512)
  const float* b_h0     = (const float*)d_in[3];  // (512)
  const float* W_in     = (const float*)d_in[4];  // (64,512)
  const float* W_rec    = (const float*)d_in[5];  // (512,512)
  const float* b_rec    = (const float*)d_in[6];  // (512)
  const float* W_out    = (const float*)d_in[7];  // (512,64)
  const float* alpha    = (const float*)d_in[8];  // (1)
  float* out = (float*)d_out;                     // (512,256,64)

  // workspace layout (f16, column-major weights): 704 KB total
  _Float16* ws = (_Float16*)d_ws;
  _Float16* Wt_h0  = ws;                      // 512*64
  _Float16* Wt_in  = ws + 512 * 64;           // 512*64
  _Float16* Wt_rec = ws + 2 * 512 * 64;       // 512*512
  _Float16* Wt_out = ws + 2 * 512 * 64 + 512 * 512;  // 64*512

  // prep: transpose + fp32->f16 (dst[n*K+k] = src[k*N+n])
  {
    int tot = 512 * 64;   // W_h0 / W_in: K=64 (log 6), N=512
    transpose_to_f16<<<(tot + 255) / 256, 256, 0, stream>>>(W_h0, Wt_h0, 6, tot, 512);
    transpose_to_f16<<<(tot + 255) / 256, 256, 0, stream>>>(W_in, Wt_in, 6, tot, 512);
  }
  {
    int tot = 512 * 512;  // W_rec: K=512 (log 9), N=512
    transpose_to_f16<<<(tot + 255) / 256, 256, 0, stream>>>(W_rec, Wt_rec, 9, tot, 512);
  }
  {
    int tot = 64 * 512;   // W_out: K=512 (log 9), N=64
    transpose_to_f16<<<(tot + 255) / 256, 256, 0, stream>>>(W_out, Wt_out, 9, tot, 64);
  }

  // main recurrence: 16 WGs x 256 threads (8 wave32s), each owns 16 batch rows
  ctrnn_kernel<<<16, 256, 0, stream>>>(h0_data, input_ts, b_h0, b_rec, alpha,
                                       Wt_h0, Wt_in, Wt_rec, Wt_out, out);
}
